// SelfAttention_61649960567323
// MI455X (gfx1250) — matrix operand order repaired and checked by reading, not compiled, not run on hardware
//
#include <hip/hip_runtime.h>
#include <hip/hip_bf16.h>

// ---------------------------------------------------------------------------
// Self-attention forward (B=2, T=2048, D=1024, H=16, HD=64) on gfx1250.
// bf16 WMMA (v_wmma_f32_16x16x32_bf16) + Tensor Data Mover (tensor_load_to_lds)
// double-buffered LDS staging. fp32 accumulate everywhere.
// ---------------------------------------------------------------------------

typedef __attribute__((ext_vector_type(16))) __bf16 v16bf;
typedef __attribute__((ext_vector_type(8)))  __bf16 v8bf;
typedef __attribute__((ext_vector_type(8)))  float  v8f;
typedef __attribute__((ext_vector_type(4)))  unsigned int u32x4;
typedef __attribute__((ext_vector_type(8)))  int i32x8;
typedef __attribute__((ext_vector_type(4)))  int i32x4;

#define Bsz 2
#define Tsz 2048
#define Dsz 1024
#define Hsz 16
#define HDsz 64

static __device__ __forceinline__ v16bf frag16(const __bf16* p0, const __bf16* p1) {
    v8bf lo, hi;
    __builtin_memcpy(&lo, p0, 16);   // ds_load_b128 / global_load_b128
    __builtin_memcpy(&hi, p1, 16);
    v16bf r;
#pragma unroll
    for (int i = 0; i < 8; ++i) { r[i] = lo[i]; r[i + 8] = hi[i]; }
    return r;
}

static __device__ __forceinline__ v8f wmma_bf16(v16bf a, v16bf b, v8f c) {
    return __builtin_amdgcn_wmma_f32_16x16x32_bf16(false, a, false, b, (short)0, c,
                                                   false, false);
}

// ---------------------------------------------------------------------------
// Tensor Data Mover: 2D bf16 tile (tile0 elems/row x tile1 rows) from global
// (row stride = stride_elems) into LDS at lds_off; pad_interval/pad_amount
// reproduce the +16B per-row LDS padding of our shared arrays.
// ---------------------------------------------------------------------------
static __device__ __forceinline__ void tdm_load_2d(
    const void* gptr, unsigned lds_off,
    unsigned stride_elems, unsigned tile0, unsigned tile1,
    unsigned padi, unsigned pada)
{
    unsigned long long ga = (unsigned long long)(size_t)gptr;
    u32x4 g0;
    g0[0] = 1u;                                            // count=1, user mode
    g0[1] = lds_off;                                       // lds_addr
    g0[2] = (unsigned)ga;                                  // global_addr[31:0]
    g0[3] = (unsigned)((ga >> 32) & 0x1FFFFFFu) | (2u << 30); // addr[56:32] | type=2
    const unsigned td0 = stride_elems;                     // tensor_dim0
    const unsigned td1 = 0x100000u;                        // tensor_dim1 (no OOB)
    i32x8 g1;
    g1[0] = (int)((1u << 16) | (1u << 20) | (padi << 22) | (pada << 25));
             // data_size=2B | pad_enable | pad_interval | pad_amount
    g1[1] = (int)((td0 & 0xFFFFu) << 16);
    g1[2] = (int)((td0 >> 16) | ((td1 & 0xFFFFu) << 16));
    g1[3] = (int)((td1 >> 16) | (tile0 << 16));            // tile_dim0
    g1[4] = (int)tile1;                                    // tile_dim1 (tile_dim2=0)
    g1[5] = (int)stride_elems;                             // tensor_dim0_stride
    g1[6] = 0;
    g1[7] = 0;
    i32x4 z4 = {0, 0, 0, 0};
#if __clang_major__ >= 23
    i32x8 z8 = {0, 0, 0, 0, 0, 0, 0, 0};
    __builtin_amdgcn_tensor_load_to_lds(g0, g1, z4, z4, z8, 0);
#else
    __builtin_amdgcn_tensor_load_to_lds(g0, g1, z4, z4, 0);
#endif
}

static __device__ __forceinline__ unsigned lds_off_of(const void* p) {
    return (unsigned)(unsigned long long)(size_t)p;        // low 32 = LDS byte addr
}

// ---------------------------------------------------------------------------
// bf16 GEMM mainloop, TDM double buffered, block tile 128(M) x 256(N),
// wave tile 64x64 (acc[4][4], 16 WMMA per K-step of 32).
// C += A(128xK) * Bt(256xK)^T ; A, Bt row-major bf16, contiguous K.
// ---------------------------------------------------------------------------
static __device__ __forceinline__ void gemm_bf16_tdm(
    const __bf16* __restrict__ A, int lda,
    const __bf16* __restrict__ Bt, int ldb,
    int m0, int n0, int K,
    __bf16 (*Ash)[128][40], __bf16 (*Bsh)[256][40],
    v8f acc[4][4])
{
    const int tid  = threadIdx.x;
    const int lane = tid & 31, wave = tid >> 5;
    const int wm = (wave >> 2) * 64, wn = (wave & 3) * 64;
    const int lh = lane & 15, half = lane >> 4;
    const int kb = half * 8, ks = half * 16;
    const int iters = K / 32;

    const unsigned ldsA[2] = { lds_off_of(&Ash[0][0][0]), lds_off_of(&Ash[1][0][0]) };
    const unsigned ldsB[2] = { lds_off_of(&Bsh[0][0][0]), lds_off_of(&Bsh[1][0][0]) };

    if (wave == 0) {
        tdm_load_2d(A + (size_t)m0 * lda, ldsA[0], lda, 32, 128, 3, 3);
        tdm_load_2d(Bt + (size_t)n0 * ldb, ldsB[0], ldb, 32, 256, 3, 3);
    }
    for (int i = 0; i < iters; ++i) {
        if (wave == 0) __builtin_amdgcn_s_wait_tensorcnt(0);
        __syncthreads();                         // tiles(i) visible to all waves
        if (wave == 0 && i + 1 < iters) {        // prefetch(i+1) overlaps compute(i)
            const int k0 = (i + 1) * 32;
            tdm_load_2d(A + (size_t)m0 * lda + k0, ldsA[(i + 1) & 1], lda, 32, 128, 3, 3);
            tdm_load_2d(Bt + (size_t)n0 * ldb + k0, ldsB[(i + 1) & 1], ldb, 32, 256, 3, 3);
        }
        __bf16 (*As)[40] = Ash[i & 1];
        __bf16 (*Bs)[40] = Bsh[i & 1];

        v16bf af[4], bfr[4];
#pragma unroll
        for (int mt = 0; mt < 4; ++mt) {
            const __bf16* p = &As[wm + mt * 16 + lh][kb];
            af[mt] = frag16(p, p + 16);
        }
#pragma unroll
        for (int nt = 0; nt < 4; ++nt) {
            const __bf16* p = &Bs[wn + nt * 16 + lh][kb];
            bfr[nt] = frag16(p, p + 16);
        }
#pragma unroll
        for (int mt = 0; mt < 4; ++mt)
#pragma unroll
            for (int nt = 0; nt < 4; ++nt)
                acc[mt][nt] = wmma_bf16(af[mt], bfr[nt], acc[mt][nt]);
    }
}

// ---------------------------------------------------------------------------
// Prep kernels: one-shot fp32 -> bf16 conversion (+ weight transpose) so the
// GEMM/attention hot loops stream bf16 and are pure TDM + WMMA.
// ---------------------------------------------------------------------------
__global__ __launch_bounds__(256) void cvt_bf16_kernel(
    const float* __restrict__ in, __bf16* __restrict__ out)
{
    const size_t i = ((size_t)blockIdx.x * 256 + threadIdx.x) * 8;
    float4 a = *(const float4*)(in + i);
    float4 b = *(const float4*)(in + i + 4);
    __bf16 o[8] = { (__bf16)a.x, (__bf16)a.y, (__bf16)a.z, (__bf16)a.w,
                    (__bf16)b.x, (__bf16)b.y, (__bf16)b.z, (__bf16)b.w };
    __builtin_memcpy(out + i, o, 16);
}

__global__ __launch_bounds__(256) void transpose_cvt_kernel(
    const float* __restrict__ in, __bf16* __restrict__ out, int rows, int cols)
{
    __shared__ float tile[32][33];
    const int c0 = blockIdx.x * 32, r0 = blockIdx.y * 32;
    const int tx = threadIdx.x & 31, ty = threadIdx.x >> 5;
#pragma unroll
    for (int r = 0; r < 4; ++r)
        tile[ty + r * 8][tx] = in[(size_t)(r0 + ty + r * 8) * cols + c0 + tx];
    __syncthreads();
#pragma unroll
    for (int r = 0; r < 4; ++r)
        out[(size_t)(c0 + ty + r * 8) * rows + r0 + tx] = (__bf16)tile[tx][ty + r * 8];
}

// ---------------------------------------------------------------------------
// Kernel 1: qkv = x @ W_attn (bf16 TDM GEMM), RoPE(q,k), 1/sqrt(HD) into q.
// q,k -> (B,H,T,HD) bf16 ; v -> (B,H,HD,T) bf16 (transposed for TDM in flash).
// grid = (3072/256, 4096/128), block 256.
// ---------------------------------------------------------------------------
__global__ __launch_bounds__(256) void qkv_rope_kernel(
    const __bf16* __restrict__ xbf, const __bf16* __restrict__ WaT,
    const float* __restrict__ rope,
    __bf16* __restrict__ qbf, __bf16* __restrict__ kbf, __bf16* __restrict__ vbf)
{
    __shared__ __bf16 Ash[2][128][40];
    __shared__ __bf16 Bsh[2][256][40];

    v8f acc[4][4] = {};
    const int m0 = blockIdx.y * 128;
    const int n0 = blockIdx.x * 256;
    gemm_bf16_tdm(xbf, Dsz, WaT, Dsz, m0, n0, Dsz, Ash, Bsh, acc);

    const int lane = threadIdx.x & 31;
    const int wave = threadIdx.x >> 5;
    const int wm = (wave >> 2) * 64, wn = (wave & 3) * 64;
    const int lh = lane & 15, half = lane >> 4;

#pragma unroll
    for (int mt = 0; mt < 4; ++mt) {
#pragma unroll
        for (int nt = 0; nt < 4; ++nt) {
            const int c   = n0 + wn + nt * 16 + lh;       // [0, 3072)
            const int sel = c >> 10;                      // 0:q 1:k 2:v
            const int cc  = c & 1023;
            const int h   = cc >> 6;
            const int hd  = cc & 63;
#pragma unroll
            for (int i = 0; i < 8; ++i) {
                const int g = m0 + wm + mt * 16 + i + half * 8;  // [0, 4096)
                const int b = g >> 11;
                const int t = g & 2047;
                float v = acc[mt][nt][i];
                float partner = __shfl_xor(v, 1, 32);     // rope pair is lane^1
                if (sel == 2) {
                    vbf[(((size_t)(b * Hsz + h)) * HDsz + hd) * Tsz + t] = (__bf16)v;
                } else {
                    const float cosv = rope[(t * 32 + (hd >> 1)) * 2 + 0];
                    const float sinv = rope[(t * 32 + (hd >> 1)) * 2 + 1];
                    float out = (hd & 1) ? (v * cosv + partner * sinv)
                                         : (v * cosv - partner * sinv);
                    if (sel == 0) out *= 0.125f;          // 1/sqrt(64)
                    __bf16* dst = sel ? kbf : qbf;
                    dst[(((size_t)(b * Hsz + h)) * Tsz + t) * HDsz + hd] = (__bf16)out;
                }
            }
        }
    }
}

// ---------------------------------------------------------------------------
// Kernel 2: causal flash attention. grid = (T/128, B*H), block 256.
// 64-key steps; K/V tiles arrive via TDM (double buffered); one barrier/step.
// Each wave owns 16 q rows.
// ---------------------------------------------------------------------------
__global__ __launch_bounds__(256) void flash_kernel(
    const __bf16* __restrict__ qbf, const __bf16* __restrict__ kbf,
    const __bf16* __restrict__ vbf, __bf16* __restrict__ abf)
{
    __shared__ __bf16 Ksh[2][64][72];    // [key][dim]
    __shared__ __bf16 Vsh[2][64][72];    // [dim][key]
    __shared__ __bf16 Psh[8][16][72];    // per-wave P staging (A layout, K=64)

    const int b  = blockIdx.y >> 4;
    const int h  = blockIdx.y & 15;
    const int qb = blockIdx.x;
    const int tid = threadIdx.x, lane = tid & 31, wave = tid >> 5;
    const int lh = lane & 15, half = lane >> 4;
    const int kb = half * 8, ks = half * 16;
    const int q0 = qb * 128 + wave * 16;
    const size_t bh  = (size_t)(b * Hsz + h) * Tsz;        // row base for q/k
    const size_t bhv = (size_t)(b * Hsz + h) * HDsz * Tsz; // base for v (HD,T)

    const unsigned ldsK[2] = { lds_off_of(&Ksh[0][0][0]), lds_off_of(&Ksh[1][0][0]) };
    const unsigned ldsV[2] = { lds_off_of(&Vsh[0][0][0]), lds_off_of(&Vsh[1][0][0]) };

    // Q fragments: straight bf16 global loads, no converts.
    v16bf qa[2];
#pragma unroll
    for (int j = 0; j < 2; ++j) {
        const __bf16* qp = qbf + (bh + q0 + lh) * HDsz + j * 32;
        qa[j] = frag16(qp + kb, qp + kb + 16);
    }

    v8f o[4] = {};
    float mrow[8], lrow[8];
#pragma unroll
    for (int i = 0; i < 8; ++i) { mrow[i] = -3.0e38f; lrow[i] = 0.0f; }

    const int nsteps = (qb + 1) * 2;                       // 64 keys per step
    if (wave == 0) {
        tdm_load_2d(kbf + bh * HDsz, ldsK[0], 64, 64, 64, 4, 3);
        tdm_load_2d(vbf + bhv,       ldsV[0], 2048, 64, 64, 4, 3);
    }
    for (int s = 0; s < nsteps; ++s) {
        const int kbase = s * 64;
        if (wave == 0) __builtin_amdgcn_s_wait_tensorcnt(0);
        __syncthreads();                                   // tiles(s) ready
        if (wave == 0 && s + 1 < nsteps) {                 // prefetch next K/V
            const int kn = (s + 1) * 64;
            tdm_load_2d(kbf + (bh + kn) * HDsz, ldsK[(s + 1) & 1], 64, 64, 64, 4, 3);
            tdm_load_2d(vbf + bhv + kn,         ldsV[(s + 1) & 1], 2048, 64, 64, 4, 3);
        }
        __bf16 (*Ks)[72] = Ksh[s & 1];
        __bf16 (*Vs)[72] = Vsh[s & 1];

        // S = Q @ K^T : four 16x16 key tiles, head dim 64 = 2 WMMAs each
        v8f sc[4] = {};
#pragma unroll
        for (int kt = 0; kt < 4; ++kt) {
            const __bf16* p0 = &Ks[kt * 16 + lh][kb];
            sc[kt] = wmma_bf16(qa[0], frag16(p0, p0 + 16), sc[kt]);
            const __bf16* p1 = &Ks[kt * 16 + lh][32 + kb];
            sc[kt] = wmma_bf16(qa[1], frag16(p1, p1 + 16), sc[kt]);
        }

        // Online softmax; row stats via butterfly shfl within 16-lane halves.
#pragma unroll
        for (int i = 0; i < 8; ++i) {
            const int qrow = q0 + i + half * 8;
            float pv[4];
            float cand = -3.0e38f;
#pragma unroll
            for (int kt = 0; kt < 4; ++kt) {
                float v = sc[kt][i];
                if (kbase + kt * 16 + lh > qrow) v = -1.0e30f;
                pv[kt] = v;
                cand = fmaxf(cand, v);
            }
            cand = fmaxf(cand, __shfl_xor(cand, 1, 32));
            cand = fmaxf(cand, __shfl_xor(cand, 2, 32));
            cand = fmaxf(cand, __shfl_xor(cand, 4, 32));
            cand = fmaxf(cand, __shfl_xor(cand, 8, 32));
            const float mnew  = fmaxf(mrow[i], cand);
            const float alpha = __expf(mrow[i] - mnew);
            float rs = 0.0f;
#pragma unroll
            for (int kt = 0; kt < 4; ++kt) { pv[kt] = __expf(pv[kt] - mnew); rs += pv[kt]; }
            rs += __shfl_xor(rs, 1, 32);
            rs += __shfl_xor(rs, 2, 32);
            rs += __shfl_xor(rs, 4, 32);
            rs += __shfl_xor(rs, 8, 32);
            lrow[i] = lrow[i] * alpha + rs;
            mrow[i] = mnew;
#pragma unroll
            for (int t = 0; t < 4; ++t) o[t][i] *= alpha;
            const int r = i + half * 8;
#pragma unroll
            for (int kt = 0; kt < 4; ++kt)
                Psh[wave][r][kt * 16 + lh] = (__bf16)pv[kt];
        }

        // Same-wave LDS RAW before re-reading P as an A fragment.
        asm volatile("s_wait_dscnt 0x0" ::: "memory");

        // O += P @ V (K = 64 keys = 2 A-fragment steps, 4 dim tiles)
#pragma unroll
        for (int j = 0; j < 2; ++j) {
            const __bf16* pp = &Psh[wave][lh][j * 32 + kb];
            const v16bf pa = frag16(pp, pp + 16);
#pragma unroll
            for (int t = 0; t < 4; ++t) {
                const __bf16* vp = &Vs[t * 16 + lh][j * 32 + kb];
                o[t] = wmma_bf16(pa, frag16(vp, vp + 16), o[t]);
            }
        }
        // No trailing barrier needed: next iteration's top barrier separates
        // all waves' compute(s) from the refill of buffer s&1 (= fill(s+2)).
    }

    // Normalize; write (B,T,D) bf16 with d = h*64 + hd (proj consumes bf16).
#pragma unroll
    for (int i = 0; i < 8; ++i) {
        const float inv = 1.0f / lrow[i];
        const int qrow = q0 + i + half * 8;
#pragma unroll
        for (int t = 0; t < 4; ++t) {
            abf[((size_t)(b * Tsz + qrow)) * Dsz + h * HDsz + t * 16 + lh] =
                (__bf16)(o[t][i] * inv);
        }
    }
}

// ---------------------------------------------------------------------------
// Kernel 3: out = attn_out @ W_proj (bf16 TDM GEMM, fp32 out).
// grid = (1024/256, 4096/128), block 256.
// ---------------------------------------------------------------------------
__global__ __launch_bounds__(256) void proj_kernel(
    const __bf16* __restrict__ Am, const __bf16* __restrict__ WpT,
    float* __restrict__ out)
{
    __shared__ __bf16 Ash[2][128][40];
    __shared__ __bf16 Bsh[2][256][40];

    v8f acc[4][4] = {};
    const int m0 = blockIdx.y * 128;
    const int n0 = blockIdx.x * 256;
    gemm_bf16_tdm(Am, Dsz, WpT, Dsz, m0, n0, Dsz, Ash, Bsh, acc);

    const int lane = threadIdx.x & 31;
    const int wave = threadIdx.x >> 5;
    const int wm = (wave >> 2) * 64, wn = (wave & 3) * 64;
    const int lh = lane & 15, half = lane >> 4;

#pragma unroll
    for (int mt = 0; mt < 4; ++mt)
#pragma unroll
        for (int nt = 0; nt < 4; ++nt) {
            const int c = n0 + wn + nt * 16 + lh;
#pragma unroll
            for (int i = 0; i < 8; ++i) {
                const int g = m0 + wm + mt * 16 + i + half * 8;
                out[(size_t)g * Dsz + c] = acc[mt][nt][i];
            }
        }
}

// ---------------------------------------------------------------------------
extern "C" void kernel_launch(void* const* d_in, const int* in_sizes, int n_in,
                              void* d_out, int out_size, void* d_ws, size_t ws_size,
                              hipStream_t stream) {
    (void)in_sizes; (void)n_in; (void)out_size; (void)ws_size;
    const float* x    = (const float*)d_in[0];   // (B,T,D)
    const float* rope = (const float*)d_in[1];   // (T,32,2)
    const float* Wa   = (const float*)d_in[2];   // (D,3D)
    const float* Wp   = (const float*)d_in[3];   // (D,D)
    float* out = (float*)d_out;

    __bf16* w = (__bf16*)d_ws;
    __bf16* xbf = w;  w += (size_t)4096 * 1024;            // x, bf16
    __bf16* WaT = w;  w += (size_t)3072 * 1024;            // W_attn^T, bf16
    __bf16* WpT = w;  w += (size_t)1024 * 1024;            // W_proj^T, bf16
    __bf16* qbf = w;  w += (size_t)Bsz * Hsz * Tsz * HDsz; // (B,H,T,HD)
    __bf16* kbf = w;  w += (size_t)Bsz * Hsz * Tsz * HDsz; // (B,H,T,HD)
    __bf16* vbf = w;  w += (size_t)Bsz * Hsz * Tsz * HDsz; // (B,H,HD,T)
    __bf16* abf = w;  w += (size_t)4096 * 1024;            // (B*T, D)

    cvt_bf16_kernel      <<<2048, 256, 0, stream>>>(x, xbf);
    transpose_cvt_kernel <<<dim3(96, 32), 256, 0, stream>>>(Wa, WaT, 1024, 3072);
    transpose_cvt_kernel <<<dim3(32, 32), 256, 0, stream>>>(Wp, WpT, 1024, 1024);
    qkv_rope_kernel      <<<dim3(12, 32), 256, 0, stream>>>(xbf, WaT, rope, qbf, kbf, vbf);
    flash_kernel         <<<dim3(16, 32), 256, 0, stream>>>(qbf, kbf, vbf, abf);
    proj_kernel          <<<dim3( 4, 32), 256, 0, stream>>>(abf, WpT, out);
}
